// GNNPortScoreFaultAware_33930241638500
// MI455X (gfx1250) — matrix operand order repaired and checked by reading, not compiled, not run on hardware
//
#include <hip/hip_runtime.h>
#include <hip/hip_bf16.h>
#include <stdint.h>

typedef _Float16 v16h __attribute__((ext_vector_type(16)));
typedef float    v8f  __attribute__((ext_vector_type(8)));

#define NN  1024
#define EE  16384
#define E2T (EE + NN)

// ---------- ordered-float encoding for atomic max ----------
__device__ __forceinline__ unsigned encf(float f) {
  unsigned u = __float_as_uint(f);
  return (u & 0x80000000u) ? ~u : (u | 0x80000000u);
}
__device__ __forceinline__ float decf(unsigned u) {
  return (u & 0x80000000u) ? __uint_as_float(u ^ 0x80000000u) : __uint_as_float(~u);
}

// ---------- small dense projection: out[n,o] = x[n,:]@W[:,o] + b[o] ----------
__global__ void k_proj(const float* __restrict__ x, const float* __restrict__ W,
                       const float* __restrict__ b, float* __restrict__ out,
                       int n, int I, int O) {
  int idx = blockIdx.x * blockDim.x + threadIdx.x;
  if (idx >= n * O) return;
  int row = idx / O, o = idx - row * O;
  float acc = b[o];
  const float* xr = x + (size_t)row * I;
  for (int k = 0; k < I; ++k) acc = fmaf(xr[k], W[(size_t)k * O + o], acc);
  out[idx] = acc;
}

// ---------- self-loop edge-attr (mean of incoming) ----------
__global__ void k_edge_stats(const int* __restrict__ dst, const float* __restrict__ ea,
                             float* __restrict__ cnt, float* __restrict__ easum) {
  int e = blockIdx.x * blockDim.x + threadIdx.x;
  if (e >= EE) return;
  int d = dst[e];
  atomicAdd(&cnt[d], 1.0f);
  atomicAdd(&easum[d], ea[e]);
}
__global__ void k_loop_fin(const float* __restrict__ cnt, const float* __restrict__ easum,
                           float* __restrict__ loop_ea) {
  int n = blockIdx.x * blockDim.x + threadIdx.x;
  if (n >= NN) return;
  loop_ea[n] = easum[n] / fmaxf(cnt[n], 1.0f);
}

// ---------- GATv2 edge passes ----------
__global__ void k_logit(const float* __restrict__ xl, const float* __restrict__ xr,
                        const float* __restrict__ We, const float* __restrict__ att,
                        const int* __restrict__ src, const int* __restrict__ dst,
                        const float* __restrict__ ea, const float* __restrict__ loop_ea,
                        float* __restrict__ logit, unsigned* __restrict__ mxenc,
                        int H, int D) {
  int idx = blockIdx.x * blockDim.x + threadIdx.x;
  if (idx >= E2T) return;
  int s, d; float eav;
  if (idx < EE) { s = src[idx]; d = dst[idx]; eav = ea[idx]; }
  else          { s = d = idx - EE; eav = loop_ea[s]; }
  int HD = H * D;
  for (int h = 0; h < H; ++h) {
    float lg = 0.f;
    for (int dd = 0; dd < D; ++dd) {
      int c = h * D + dd;
      float m = xl[(size_t)s * HD + c] + xr[(size_t)d * HD + c] + eav * We[c];
      m = fmaxf(m, 0.2f * m);            // leaky_relu(0.2)
      lg = fmaf(att[c], m, lg);
    }
    logit[(size_t)idx * H + h] = lg;
    atomicMax(&mxenc[d * H + h], encf(lg));
  }
}

__global__ void k_expsum(const float* __restrict__ logit, const int* __restrict__ dst,
                         const unsigned* __restrict__ mxenc,
                         float* __restrict__ exb, float* __restrict__ den, int H) {
  int idx = blockIdx.x * blockDim.x + threadIdx.x;
  if (idx >= E2T) return;
  int d = (idx < EE) ? dst[idx] : idx - EE;
  for (int h = 0; h < H; ++h) {
    float ex = expf(logit[(size_t)idx * H + h] - decf(mxenc[d * H + h]));
    exb[(size_t)idx * H + h] = ex;
    atomicAdd(&den[d * H + h], ex);
  }
}

__global__ void k_accum(const float* __restrict__ xl, const float* __restrict__ exb,
                        const float* __restrict__ den,
                        const int* __restrict__ src, const int* __restrict__ dst,
                        float* __restrict__ acc, int H, int D) {
  int idx = blockIdx.x * blockDim.x + threadIdx.x;
  if (idx >= E2T) return;
  int s, d;
  if (idx < EE) { s = src[idx]; d = dst[idx]; } else { s = d = idx - EE; }
  int HD = H * D;
  for (int h = 0; h < H; ++h) {
    float alpha = exb[(size_t)idx * H + h] / den[d * H + h];
    for (int dd = 0; dd < D; ++dd) {
      int c = h * D + dd;
      atomicAdd(&acc[(size_t)d * HD + c], alpha * xl[(size_t)s * HD + c]);
    }
  }
}

// ---------- per-node LayerNorm (+ optional ELU) ----------
__global__ void k_ln_act(const float* __restrict__ acc, const float* __restrict__ bias,
                         const float* __restrict__ g, const float* __restrict__ b,
                         float* __restrict__ out, int HD, int do_elu) {
  int n = blockIdx.x * blockDim.x + threadIdx.x;
  if (n >= NN) return;
  float s1 = 0.f, s2 = 0.f;
  for (int c = 0; c < HD; ++c) {
    float v = acc[(size_t)n * HD + c] + bias[c];
    s1 += v; s2 += v * v;
  }
  float mu  = s1 / (float)HD;
  float var = fmaxf(s2 / (float)HD - mu * mu, 0.f);
  float inv = rsqrtf(var + 1e-5f);
  for (int c = 0; c < HD; ++c) {
    float v = acc[(size_t)n * HD + c] + bias[c];
    float y = (v - mu) * inv * g[c] + b[c];
    if (do_elu) y = (y > 0.f) ? y : expm1f(y);
    out[(size_t)n * HD + c] = y;
  }
}

// ---------- decoder precompute: u = emb@W1[:32]+b1, v = emb@W1[32:] ----------
__global__ void k_dec_pre(const float* __restrict__ emb,
                          const float* __restrict__ W1, const float* __restrict__ b1,
                          float* __restrict__ uf, float* __restrict__ vf,
                          _Float16* __restrict__ u16, _Float16* __restrict__ v16,
                          float* __restrict__ su, float* __restrict__ qu,
                          float* __restrict__ sv, float* __restrict__ qv) {
  int idx = blockIdx.x * blockDim.x + threadIdx.x;
  if (idx >= 4 * NN) return;
  int p = idx >> 10, n = idx & (NN - 1);
  float e[32];
#pragma unroll
  for (int k = 0; k < 32; ++k) e[k] = emb[(size_t)n * 32 + k];
  float s_u = 0, q_u = 0, s_v = 0, q_v = 0;
  for (int c = 0; c < 32; ++c) {
    float u = b1[p * 32 + c];
    float v = 0.f;
#pragma unroll
    for (int k = 0; k < 32; ++k) {
      u = fmaf(e[k], W1[(size_t)(p * 64 + k) * 32 + c], u);
      v = fmaf(e[k], W1[(size_t)(p * 64 + 32 + k) * 32 + c], v);
    }
    size_t o = (size_t)((p << 10) + n) * 32 + c;
    uf[o] = u; vf[o] = v;
    u16[o] = (_Float16)u; v16[o] = (_Float16)v;
    s_u += u; q_u += u * u; s_v += v; q_v += v * v;
  }
  su[(p << 10) + n] = s_u * (1.f / 32.f);
  qu[(p << 10) + n] = q_u * (1.f / 32.f);
  sv[(p << 10) + n] = s_v * (1.f / 32.f);
  qv[(p << 10) + n] = q_v * (1.f / 32.f);
}

// ---------- main decoder: WMMA cross-dot + fused LN/leakyReLU/dot epilogue ----------
// 128 threads = 4 waves; block tile = 16(i) x 64(j); all LDS staged once.
__global__ __launch_bounds__(128) void k_decode(
    const _Float16* __restrict__ u16, const _Float16* __restrict__ v16,
    const float* __restrict__ uf, const float* __restrict__ vf,
    const float* __restrict__ su, const float* __restrict__ qu,
    const float* __restrict__ sv, const float* __restrict__ qv,
    const float* __restrict__ dg, const float* __restrict__ dbt,
    const float* __restrict__ dW2, const float* __restrict__ db2,
    float* __restrict__ out) {
  __shared__ float ufs[4][16][33];        // per-port u rows of the i-tile (padded)
  __shared__ float vfs[4][4][16][33];     // per-wave, per-port v rows (padded)
  __shared__ float sus[4][16], qus[4][16];

  const int tid  = threadIdx.x;
  const int wave = tid >> 5;
  const int lane = tid & 31;
  const int ln   = lane & 15;
  const int hi   = lane >> 4;
  const int i0   = blockIdx.y * 16;
  const int j0   = blockIdx.x * 64 + wave * 16;
  const int jcol = j0 + ln;

  // stage all u tiles (4 ports x 16 x 32 = 2048 floats; 16 per thread)
  for (int e = tid; e < 2048; e += 128) {
    int p = e >> 9, rem = e & 511, row = rem >> 5, c = rem & 31;
    ufs[p][row][c] = uf[(size_t)((p << 10) + i0 + row) * 32 + c];
  }
  if (tid < 64) {
    int p = tid >> 4, r = tid & 15;
    sus[p][r] = su[(p << 10) + i0 + r];
    qus[p][r] = qu[(p << 10) + i0 + r];
  }
  // stage this wave's v tiles (4 ports x 16 x 32 = 2048 floats; 64 per lane)
  for (int t = lane; t < 2048; t += 32) {
    int p = t >> 9, rem = t & 511, row = rem >> 5, c = rem & 31;
    vfs[wave][p][row][c] = vf[(size_t)((p << 10) + j0 + row) * 32 + c];
  }
  __syncthreads();

  float sc[8][4];

#pragma unroll
  for (int p = 0; p < 4; ++p) {
    // WMMA fragments per ISA f16 layouts
    union { v16h v; uint4 q[2]; } A, B;
    {
      const _Float16* ap = u16 + (size_t)((p << 10) + i0 + ln) * 32;
      const int kbA = hi ? 8 : 0;                 // A: K {kb..kb+7, 16+kb..16+kb+7}
      A.q[0] = *(const uint4*)(ap + kbA);
      A.q[1] = *(const uint4*)(ap + 16 + kbA);
      const _Float16* bp = v16 + (size_t)((p << 10) + jcol) * 32 + kbA;
      B.q[0] = *(const uint4*)(bp);               // B: K {kb..kb+7, 16+kb..16+kb+7}
      B.q[1] = *(const uint4*)(bp + 16);
    }
    v8f c8 = {};
    c8 = __builtin_amdgcn_wmma_f32_16x16x32_f16(false, A.v, false, B.v,
                                                (short)0, c8, false, false);

    const float svj = sv[(p << 10) + jcol];
    const float qvj = qv[(p << 10) + jcol];
    const float b2p = db2[p];

    float mu[8], inv[8], acc[8];
#pragma unroll
    for (int r = 0; r < 8; ++r) {
      const int m = r + hi * 8;                        // D layout: M = r + 8*(lane>=16)
      mu[r] = sus[p][m] + svj;
      float ms = qus[p][m] + qvj + c8[r] * (2.0f / 32.0f);
      inv[r] = rsqrtf(fmaxf(ms - mu[r] * mu[r], 0.0f) + 1e-5f);
      acc[r] = b2p;
    }

    const float* gp   = dg  + p * 32;
    const float* btp  = dbt + p * 32;
    const float* w2p  = dW2 + p * 32;
    const float* vrow = &vfs[wave][p][ln][0];
    const float* ub   = &ufs[p][hi * 8][0];            // 8 rows, stride 33

#pragma clang loop unroll(disable)
    for (int cc = 0; cc < 32; ++cc) {
      const float vcc = vrow[cc];
      const float gc  = gp[cc];
      const float btc = btp[cc];
      const float w2c = w2p[cc];
#pragma unroll
      for (int r = 0; r < 8; ++r) {
        float h = ub[r * 33 + cc] + vcc;
        float y = fmaf((h - mu[r]) * inv[r], gc, btc);
        acc[r] = fmaf(w2c, fmaxf(y, 0.1f * y), acc[r]); // leaky_relu(0.1)
      }
    }
#pragma unroll
    for (int r = 0; r < 8; ++r) sc[r][p] = acc[r];
  }

#pragma unroll
  for (int r = 0; r < 8; ++r) {
    const int irow = i0 + r + hi * 8;
    float4 o4 = make_float4(sc[r][0], sc[r][1], sc[r][2], sc[r][3]);
    *(float4*)(out + ((size_t)irow * NN + jcol) * 4) = o4;   // b128 store
  }
}

// ---------------------------------------------------------------------------
extern "C" void kernel_launch(void* const* d_in, const int* in_sizes, int n_in,
                              void* d_out, int out_size, void* d_ws, size_t ws_size,
                              hipStream_t stream) {
  (void)in_sizes; (void)n_in; (void)out_size; (void)ws_size;
  const float* x   = (const float*)d_in[0];
  const int*   ei  = (const int*)d_in[1];
  const float* ea  = (const float*)d_in[2];
  const int* srcp = ei;
  const int* dstp = ei + EE;

  struct Conv { const float *Wl, *bl, *Wr, *br, *We, *att, *bias; };
  auto getc = [&](int b) {
    Conv c = {(const float*)d_in[b],   (const float*)d_in[b+1], (const float*)d_in[b+2],
              (const float*)d_in[b+3], (const float*)d_in[b+4], (const float*)d_in[b+5],
              (const float*)d_in[b+6]};
    return c;
  };
  Conv c1 = getc(3), c2 = getc(10), c3 = getc(17);
  const float* ln1g = (const float*)d_in[24]; const float* ln1b = (const float*)d_in[25];
  const float* ln2g = (const float*)d_in[26]; const float* ln2b = (const float*)d_in[27];
  const float* ln3g = (const float*)d_in[28]; const float* ln3b = (const float*)d_in[29];
  const float* dW1  = (const float*)d_in[30]; const float* db1  = (const float*)d_in[31];
  const float* ddg  = (const float*)d_in[32]; const float* ddbt = (const float*)d_in[33];
  const float* ddW2 = (const float*)d_in[34]; const float* ddb2 = (const float*)d_in[35];

  char* w = (char*)d_ws; size_t off = 0;
  auto alloc = [&](size_t bytes) -> void* {
    void* p = w + off; off = (off + bytes + 255) & ~(size_t)255; return p;
  };
  float*    xl      = (float*)alloc((size_t)NN * 64 * 4);
  float*    xr      = (float*)alloc((size_t)NN * 64 * 4);
  float*    accum   = (float*)alloc((size_t)NN * 64 * 4);
  float*    hA      = (float*)alloc((size_t)NN * 64 * 4);
  float*    hB      = (float*)alloc((size_t)NN * 64 * 4);
  float*    emb     = (float*)alloc((size_t)NN * 32 * 4);
  float*    logit   = (float*)alloc((size_t)E2T * 4 * 4);
  float*    exb     = (float*)alloc((size_t)E2T * 4 * 4);
  unsigned* mxenc   = (unsigned*)alloc((size_t)NN * 4 * 4);
  float*    den     = (float*)alloc((size_t)NN * 4 * 4);
  float*    cnt     = (float*)alloc((size_t)NN * 4);
  float*    easum   = (float*)alloc((size_t)NN * 4);
  float*    loop_ea = (float*)alloc((size_t)NN * 4);
  float*    uf      = (float*)alloc((size_t)4 * NN * 32 * 4);
  float*    vf      = (float*)alloc((size_t)4 * NN * 32 * 4);
  float*    su      = (float*)alloc((size_t)4 * NN * 4);
  float*    qu      = (float*)alloc((size_t)4 * NN * 4);
  float*    sv      = (float*)alloc((size_t)4 * NN * 4);
  float*    qv      = (float*)alloc((size_t)4 * NN * 4);
  _Float16* u16     = (_Float16*)alloc((size_t)4 * NN * 32 * 2);
  _Float16* v16     = (_Float16*)alloc((size_t)4 * NN * 32 * 2);

  const int TB = 256;
  // self-loop edge attrs
  hipMemsetAsync(cnt,   0, (size_t)NN * 4, stream);
  hipMemsetAsync(easum, 0, (size_t)NN * 4, stream);
  k_edge_stats<<<(EE + TB - 1) / TB, TB, 0, stream>>>(dstp, ea, cnt, easum);
  k_loop_fin<<<(NN + TB - 1) / TB, TB, 0, stream>>>(cnt, easum, loop_ea);

  auto gat = [&](const float* hin, int I, int H, int D, const Conv& cp,
                 const float* lg, const float* lb, int elu, float* hout) {
    int HD = H * D;
    hipMemsetAsync(accum, 0, (size_t)NN * HD * 4, stream);
    hipMemsetAsync(mxenc, 0, (size_t)NN * H * 4, stream);
    hipMemsetAsync(den,   0, (size_t)NN * H * 4, stream);
    int g1 = (NN * HD + TB - 1) / TB;
    k_proj<<<g1, TB, 0, stream>>>(hin, cp.Wl, cp.bl, xl, NN, I, HD);
    k_proj<<<g1, TB, 0, stream>>>(hin, cp.Wr, cp.br, xr, NN, I, HD);
    int ge = (E2T + TB - 1) / TB;
    k_logit<<<ge, TB, 0, stream>>>(xl, xr, cp.We, cp.att, srcp, dstp, ea, loop_ea,
                                   logit, mxenc, H, D);
    k_expsum<<<ge, TB, 0, stream>>>(logit, dstp, mxenc, exb, den, H);
    k_accum<<<ge, TB, 0, stream>>>(xl, exb, den, srcp, dstp, accum, H, D);
    k_ln_act<<<(NN + TB - 1) / TB, TB, 0, stream>>>(accum, cp.bias, lg, lb, hout, HD, elu);
  };
  gat(x,  12, 4, 16, c1, ln1g, ln1b, 1, hA);
  gat(hA, 64, 4, 16, c2, ln2g, ln2b, 1, hB);
  gat(hB, 64, 1, 32, c3, ln3g, ln3b, 0, emb);

  k_dec_pre<<<(4 * NN + TB - 1) / TB, TB, 0, stream>>>(emb, dW1, db1, uf, vf, u16, v16,
                                                       su, qu, sv, qv);
  dim3 grid(NN / 64, NN / 16);
  k_decode<<<grid, 128, 0, stream>>>(u16, v16, uf, vf, su, qu, sv, qv,
                                     ddg, ddbt, ddW2, ddb2, (float*)d_out);
}
